// QuantizedSelectiveSSMBlock_68032281969133
// MI455X (gfx1250) — matrix operand order repaired and checked by reading, not compiled, not run on hardware
//
#include <hip/hip_runtime.h>
#include <hip/hip_bf16.h>
#include <math.h>

#define DM    1024
#define DI    2048
#define DST   16
#define DCONV 4
#define BATCH 4
#define SEQ   2048
#define NTOK  (BATCH*SEQ)
#define NPROJ (2*DST+1)   /* 33 */

typedef __bf16  bf16_t;
typedef bf16_t  v16bf __attribute__((ext_vector_type(16)));
typedef float   v8f   __attribute__((ext_vector_type(8)));
typedef int     i32x4_ __attribute__((ext_vector_type(4)));
typedef int     i32x2_ __attribute__((ext_vector_type(2)));

#if defined(__has_builtin)
#if __has_builtin(__builtin_amdgcn_global_load_async_to_lds_b128) && \
    __has_builtin(__builtin_amdgcn_global_load_async_to_lds_b64) &&  \
    __has_builtin(__builtin_amdgcn_s_wait_asynccnt)
#define USE_ASYNC_LDS 1
#endif
#endif

#define ASG __attribute__((address_space(1)))
#define ASL __attribute__((address_space(3)))

__device__ __forceinline__ unsigned short f2bf(float f){
  unsigned int u = __float_as_uint(f);
  u += 0x7FFFu + ((u >> 16) & 1u);          // round-to-nearest-even
  return (unsigned short)(u >> 16);
}
__device__ __forceinline__ bf16_t us2bf(unsigned short s){
  union { unsigned short u; bf16_t b; } x; x.u = s; return x.b;
}
__device__ __forceinline__ float sigmoidf_(float v){ return 1.0f/(1.0f+__expf(-v)); }
__device__ __forceinline__ float softplusf_(float v){ return v > 20.0f ? v : log1pf(__expf(v)); }

// ---------------------------------------------------------------- f32 -> bf16
__global__ void k_f32_to_bf16(const float* __restrict__ src,
                              unsigned short* __restrict__ dst, int n){
  int i = blockIdx.x*blockDim.x + threadIdx.x;
  if (i < n) dst[i] = f2bf(src[i]);
}

// ---------------------------------------------------------------- LayerNorm
__global__ void __launch_bounds__(256)
k_layernorm_bf16(const float* __restrict__ x, const float* __restrict__ w,
                 const float* __restrict__ b, unsigned short* __restrict__ xn){
  __shared__ float lsS[8], lsQ[8];
  int tok = blockIdx.x;
  int tid = threadIdx.x;
  const float* row = x + (size_t)tok*DM;
  float v[4]; float s = 0.f, q = 0.f;
  #pragma unroll
  for (int i=0;i<4;i++){ v[i] = row[tid + i*256]; s += v[i]; q += v[i]*v[i]; }
  #pragma unroll
  for (int off=16; off>0; off>>=1){ s += __shfl_xor(s, off, 32); q += __shfl_xor(q, off, 32); }
  int lane = tid & 31, wv = tid >> 5;
  if (lane==0){ lsS[wv]=s; lsQ[wv]=q; }
  __syncthreads();
  if (tid==0){
    float ts=0.f, tq=0.f;
    #pragma unroll
    for (int i=0;i<8;i++){ ts += lsS[i]; tq += lsQ[i]; }
    float mu  = ts / (float)DM;
    float var = tq / (float)DM - mu*mu;
    lsS[0]=mu; lsQ[0]=rsqrtf(var + 1e-5f);
  }
  __syncthreads();
  float mu = lsS[0], rstd = lsQ[0];
  #pragma unroll
  for (int i=0;i<4;i++){
    int c = tid + i*256;
    xn[(size_t)tok*DM + c] = f2bf((v[i]-mu)*rstd*w[c] + b[c]);
  }
}

// ------------------------------------------------- bf16 WMMA GEMM  C = A*W^T
// A: (M,K) bf16 row-major.  W: (N,K) bf16 row-major.  out = acc + bias[ +resid]
// block = 256 thr (8 waves), tile 64M x 32N, K-step 32, double-buffered
// async global->LDS staging (GLOBAL_LOAD_ASYNC_TO_LDS_*, ASYNCcnt).
__global__ void __launch_bounds__(256)
k_gemm_bf16_wmma(const unsigned short* __restrict__ A,
                 const unsigned short* __restrict__ W,
                 const float* __restrict__ bias,
                 const float* __restrict__ resid,
                 float* __restrict__ out, int M, int N, int K){
#ifdef USE_ASYNC_LDS
  __shared__ unsigned short lsA[2][64*32];
  __shared__ unsigned short lsB[2][32*32];
#else
  __shared__ unsigned short lsA[1][64*32];
  __shared__ unsigned short lsB[1][32*32];
#endif
  int tid  = threadIdx.x;
  int lane = tid & 31, wave = tid >> 5;
  int wm = wave >> 1, wn = wave & 1;
  int lhalf = (lane >> 4);                  // 0: lanes 0-15, 1: lanes 16-31
  int m0 = blockIdx.y * 64;
  int n0 = blockIdx.x * 32;

  v8f acc = {0.f,0.f,0.f,0.f,0.f,0.f,0.f,0.f};

  // per-thread staging slots
  int rA = tid >> 2,  cA = (tid & 3) * 8;   // A tile: one b128 per thread
  int rB = tid >> 3,  cB = (tid & 7) * 4;   // B tile: one b64  per thread
  const unsigned short* gA = A + (size_t)(m0 + rA)*K + cA;
  const unsigned short* gW = W + (size_t)(n0 + rB)*K + cB;

  int rowA = wm*16 + (lane & 15);
  int rowB = wn*16 + (lane & 15);

#ifdef USE_ASYNC_LDS
  auto issue = [&](int k0, int buf){
    __builtin_amdgcn_global_load_async_to_lds_b128(
        (ASG i32x4_*)(gA + k0),
        (ASL i32x4_*)&lsA[buf][rA*32 + cA], 0, 0);
    __builtin_amdgcn_global_load_async_to_lds_b64(
        (ASG i32x2_*)(gW + k0),
        (ASL i32x2_*)&lsB[buf][rB*32 + cB], 0, 0);
  };
  int nk = K >> 5;
  issue(0, 0);
  for (int t = 0; t < nk; ++t){
    int buf = t & 1;
    if (t + 1 < nk){
      issue((t+1) << 5, buf ^ 1);           // prefetch next tile into other buf
      __builtin_amdgcn_s_wait_asynccnt(2);  // previous pair complete (in-order)
    } else {
      __builtin_amdgcn_s_wait_asynccnt(0);
    }
    __syncthreads();
    // build fragments per CDNA5 ISA layout (CDNA5 ISA §7.12.2)
    v16bf a, b;
    #pragma unroll
    for (int i=0;i<16;i++){
      int ka = i + (i & 8) + (lhalf << 3);  // K = i + 8*(i>=8) + 8*(lane>=16)
      a[i] = us2bf(lsA[buf][rowA*32 + ka]);
      int kb = i + (i & 8) + (lhalf << 3);  // K = i + 8*(i>=8) + 8*(lane>=16)
      b[i] = us2bf(lsB[buf][rowB*32 + kb]);
    }
    acc = __builtin_amdgcn_wmma_f32_16x16x32_bf16(
              false, a, false, b, (short)0, acc, false, false);
    __syncthreads();
  }
#else
  for (int k0 = 0; k0 < K; k0 += 32){
    *(uint4*)&lsA[0][rA*32 + cA] = *(const uint4*)(gA + k0);
    *(uint2*)&lsB[0][rB*32 + cB] = *(const uint2*)(gW + k0);
    if (k0 + 32 < K)
      __builtin_prefetch(gA + k0 + 32, 0, 3);
    __syncthreads();
    v16bf a, b;
    #pragma unroll
    for (int i=0;i<16;i++){
      int ka = i + (i & 8) + (lhalf << 3);
      a[i] = us2bf(lsA[0][rowA*32 + ka]);
      int kb = i + (i & 8) + (lhalf << 3);
      b[i] = us2bf(lsB[0][rowB*32 + kb]);
    }
    acc = __builtin_amdgcn_wmma_f32_16x16x32_bf16(
              false, a, false, b, (short)0, acc, false, false);
    __syncthreads();
  }
#endif

  int gn = n0 + wn*16 + (lane & 15);
  #pragma unroll
  for (int v=0; v<8; v++){                  // D: M = v + 8*(lane>=16), N = lane&15
    int gm = m0 + wm*16 + v + (lhalf << 3);
    float r = acc[v] + bias[gn];
    if (resid) r += resid[(size_t)gm*N + gn];
    out[(size_t)gm*N + gn] = r;
  }
}

// ---------------------------------------------------- depthwise conv + SiLU
__global__ void k_conv_silu(const float* __restrict__ xz, const float* __restrict__ cw,
                            const float* __restrict__ cb, float* __restrict__ xconv){
  int idx = blockIdx.x*blockDim.x + threadIdx.x;
  if (idx >= NTOK*DI) return;
  int c = idx % DI;
  int tok = idx / DI;
  int l = tok % SEQ;
  int b = tok / SEQ;
  float s = cb[c];
  #pragma unroll
  for (int j=0;j<DCONV;j++){
    int lp = l - (DCONV-1) + j;
    if (lp >= 0) s += xz[((size_t)(b*SEQ + lp))*(2*DI) + c] * cw[c*DCONV + j];
  }
  xconv[idx] = s * sigmoidf_(s);
}

// ------------------------------------------------- x_proj: 33 dots per token
__global__ void __launch_bounds__(256)
k_xproj(const float* __restrict__ xconv, const float* __restrict__ w,
        const float* __restrict__ b, float* __restrict__ BCbuf,
        float* __restrict__ sdt){
  __shared__ float row[DI];
  int tok = blockIdx.x, tid = threadIdx.x;
  for (int i = tid; i < DI; i += 256) row[i] = xconv[(size_t)tok*DI + i];
  __syncthreads();
  int lane = tid & 31, wave = tid >> 5;
  for (int o = wave; o < NPROJ; o += 8){
    const float* wr = w + (size_t)o*DI;
    float s = 0.f;
    for (int k = lane; k < DI; k += 32) s += wr[k]*row[k];
    #pragma unroll
    for (int off=16; off>0; off>>=1) s += __shfl_xor(s, off, 32);
    if (lane==0){
      s += b[o];
      if (o < 2*DST) BCbuf[(size_t)tok*32 + o] = s;
      else           sdt[tok] = s;
    }
  }
}

// ------------------------ sequential scan with per-step quant-dequant (norm)
// one block per batch; 1024 thr; thread owns 2 channels x 16 states in regs
__global__ void __launch_bounds__(1024)
k_scan(const float* __restrict__ xconv, const float* __restrict__ BCbuf,
       const float* __restrict__ sdt, const float* __restrict__ dtw,
       const float* __restrict__ dtb, const float* __restrict__ Alog,
       float* __restrict__ y){
  __shared__ float lsBC[32];
  __shared__ float lsR[32];
  int b = blockIdx.x;
  int tid = threadIdx.x;
  int lane = tid & 31, wave = tid >> 5;
  const float Q_STEP = 4.0f / (sqrtf((float)(DI*DST)) * 7.0f);
  float aS[DST];
  #pragma unroll
  for (int s=0;s<DST;s++) aS[s] = -__expf(Alog[s]);
  int c0 = tid*2, c1 = tid*2 + 1;
  float dw0 = dtw[c0], dw1 = dtw[c1];
  float db0 = dtb[c0], db1 = dtb[c1];
  float h0[DST], h1[DST];
  #pragma unroll
  for (int s=0;s<DST;s++){ h0[s]=0.f; h1[s]=0.f; }

  for (int l=0; l<SEQ; l++){
    int tok = b*SEQ + l;
    if (tid < 32) lsBC[tid] = BCbuf[(size_t)tok*32 + tid];
    __syncthreads();
    float sv  = sdt[tok];
    float dt0 = softplusf_(sv*dw0 + db0);
    float dt1 = softplusf_(sv*dw1 + db1);
    float x0  = xconv[(size_t)tok*DI + c0];
    float x1  = xconv[(size_t)tok*DI + c1];
    float ss = 0.f;
    #pragma unroll
    for (int s=0;s<DST;s++){
      float Bv = lsBC[s];
      h0[s] = __expf(aS[s]*dt0)*h0[s] + dt0*Bv*x0;
      h1[s] = __expf(aS[s]*dt1)*h1[s] + dt1*Bv*x1;
      ss += h0[s]*h0[s] + h1[s]*h1[s];
    }
    #pragma unroll
    for (int off=16; off>0; off>>=1) ss += __shfl_xor(ss, off, 32);
    if (lane==0) lsR[wave] = ss;
    __syncthreads();
    if (tid < 32){
      float v = lsR[tid];
      #pragma unroll
      for (int off=16; off>0; off>>=1) v += __shfl_xor(v, off, 32);
      if (tid==0) lsR[0] = v;
    }
    __syncthreads();
    float scale = sqrtf(lsR[0]);
    float inv   = 1.0f / fmaxf(scale, 1e-8f);
    float y0 = 0.f, y1 = 0.f;
    #pragma unroll
    for (int s=0;s<DST;s++){
      float q0 = fminf(fmaxf(rintf(h0[s]*inv/Q_STEP), -7.f), 7.f);
      float q1 = fminf(fmaxf(rintf(h1[s]*inv/Q_STEP), -7.f), 7.f);
      h0[s] = q0*Q_STEP*scale;
      h1[s] = q1*Q_STEP*scale;
      float Cv = lsBC[DST+s];
      y0 += h0[s]*Cv;
      y1 += h1[s]*Cv;
    }
    y[(size_t)tok*DI + c0] = y0;
    y[(size_t)tok*DI + c1] = y1;
    __syncthreads();
  }
}

// --------------------------------------------- gating: (y + D*xc)*silu(z)
__global__ void k_gate_bf16(const float* __restrict__ y, const float* __restrict__ xconv,
                            const float* __restrict__ Dw, const float* __restrict__ xz,
                            unsigned short* __restrict__ y2){
  int idx = blockIdx.x*blockDim.x + threadIdx.x;
  if (idx >= NTOK*DI) return;
  int c = idx % DI;
  int tok = idx / DI;
  float z = xz[(size_t)tok*(2*DI) + DI + c];
  float v = (y[idx] + Dw[c]*xconv[idx]) * (z * sigmoidf_(z));
  y2[idx] = f2bf(v);
}

// ---------------------------------------------------------------- launcher
extern "C" void kernel_launch(void* const* d_in, const int* in_sizes, int n_in,
                              void* d_out, int out_size, void* d_ws, size_t ws_size,
                              hipStream_t stream){
  const float* x        = (const float*)d_in[0];
  const float* norm_w   = (const float*)d_in[1];
  const float* norm_b   = (const float*)d_in[2];
  const float* in_w     = (const float*)d_in[3];
  const float* in_b     = (const float*)d_in[4];
  const float* conv_w   = (const float*)d_in[5];
  const float* conv_b   = (const float*)d_in[6];
  const float* xp_w     = (const float*)d_in[7];
  const float* xp_b     = (const float*)d_in[8];
  const float* dt_w     = (const float*)d_in[9];
  const float* dt_b     = (const float*)d_in[10];
  const float* A_log    = (const float*)d_in[11];
  const float* Dw       = (const float*)d_in[12];
  const float* out_w    = (const float*)d_in[13];
  const float* out_b    = (const float*)d_in[14];
  float* out = (float*)d_out;

  char* ws = (char*)d_ws;
  size_t o = 0;
  auto alloc = [&](size_t bytes)->size_t{
    size_t r = o; o = (o + bytes + 255) & ~(size_t)255; return r;
  };
  unsigned short* w1b  = (unsigned short*)(ws + alloc((size_t)2*DI*DM*2));   // 8.4 MB
  unsigned short* w2b  = (unsigned short*)(ws + alloc((size_t)DM*DI*2));     // 4.2 MB
  unsigned short* xnb  = (unsigned short*)(ws + alloc((size_t)NTOK*DM*2));   // 16.8 MB
  float*          xz   = (float*)(ws + alloc((size_t)NTOK*2*DI*4));          // 134 MB
  float*          xcv  = (float*)(ws + alloc((size_t)NTOK*DI*4));            // 67 MB
  float*          BCb  = (float*)(ws + alloc((size_t)NTOK*32*4));            // 1 MB
  float*          sdt  = (float*)(ws + alloc((size_t)NTOK*4));               // 32 KB
  float*          ybuf = (float*)(ws + alloc((size_t)NTOK*DI*4));            // 67 MB
  unsigned short* y2b  = (unsigned short*)(ws + alloc((size_t)NTOK*DI*2));   // 33.5 MB
  (void)ws_size; (void)in_sizes; (void)n_in; (void)out_size;

  // weight conversion to bf16
  {
    int n1 = 2*DI*DM, n2 = DM*DI;
    k_f32_to_bf16<<<(n1+255)/256, 256, 0, stream>>>(in_w,  w1b, n1);
    k_f32_to_bf16<<<(n2+255)/256, 256, 0, stream>>>(out_w, w2b, n2);
  }
  // LayerNorm -> bf16 activations
  k_layernorm_bf16<<<NTOK, 256, 0, stream>>>(x, norm_w, norm_b, xnb);
  // in_proj GEMM: (8192,1024) x (4096,1024)^T -> xz (8192,4096)
  {
    dim3 g((2*DI)/32, NTOK/64);
    k_gemm_bf16_wmma<<<g, 256, 0, stream>>>(xnb, w1b, in_b, nullptr, xz,
                                            NTOK, 2*DI, DM);
  }
  // depthwise causal conv + SiLU
  k_conv_silu<<<(NTOK*DI)/256, 256, 0, stream>>>(xz, conv_w, conv_b, xcv);
  // x_proj (B, C, dt-scalar)
  k_xproj<<<NTOK, 256, 0, stream>>>(xcv, xp_w, xp_b, BCb, sdt);
  // sequential selective scan with per-step quant-dequant
  k_scan<<<BATCH, 1024, 0, stream>>>(xcv, BCb, sdt, dt_w, dt_b, A_log, ybuf);
  // gating -> bf16
  k_gate_bf16<<<(NTOK*DI)/256, 256, 0, stream>>>(ybuf, xcv, Dw, xz, y2b);
  // out_proj GEMM + bias + residual: (8192,2048) x (1024,2048)^T -> out
  {
    dim3 g(DM/32, NTOK/64);
    k_gemm_bf16_wmma<<<g, 256, 0, stream>>>(y2b, w2b, out_b, x, out,
                                            NTOK, DM, DI);
  }
}
